// LayerNormNeuralCDE_850403525417
// MI455X (gfx1250) — matrix operand order repaired and checked by reading, not compiled, not run on hardware
//
#include <hip/hip_runtime.h>
#include <hip/hip_bf16.h>
#include <math.h>
#include <stdint.h>

// Problem dims (from reference)
#define BATCH 1024
#define SEQ   32
#define DIN   64
#define DH    128
#define DH2   256
#define DHI   8192   // H*IN
#define NTRAITS 20
#define NSUB  2      // RK4 substeps per interval

typedef __attribute__((ext_vector_type(16))) __bf16 bf16x16;
typedef __attribute__((ext_vector_type(8)))  float  f32x8;

// ---------- scalar helpers ----------
static __device__ __forceinline__ __bf16 f2bf(float f) {
  unsigned u = __float_as_uint(f);
  unsigned r = u + 0x7FFFu + ((u >> 16) & 1u);   // round-to-nearest-even
  unsigned short h = (unsigned short)(r >> 16);
  __bf16 o; __builtin_memcpy(&o, &h, 2); return o;
}
static __device__ __forceinline__ float bf2f(__bf16 b) {
  unsigned short h; __builtin_memcpy(&h, &b, 2);
  return __uint_as_float(((unsigned)h) << 16);
}
static __device__ __forceinline__ float gelu_erf(float x) {
  return 0.5f * x * (1.0f + erff(x * 0.70710678118654752440f));
}
// xor-lane add via ds_swizzle_b32 (group-of-32: new_lane = (lane & 0x1f) ^ M).
// EXEC is all-ones wherever this is used; avoids __shfl's bounds cndmask chains.
template <int M>
static __device__ __forceinline__ float swz_add(float x) {
  int y = __builtin_amdgcn_ds_swizzle(__float_as_int(x), (M << 10) | 0x1f);
  return x + __int_as_float(y);
}
static __device__ __forceinline__ float red16(float x) {  // reduce within 16-lane half
  x = swz_add<8>(x); x = swz_add<4>(x); x = swz_add<2>(x); x = swz_add<1>(x); return x;
}
static __device__ __forceinline__ float red32(float x) {  // reduce across full wave
  x = swz_add<16>(x); return red16(x);
}

// ---------- WMMA helpers (CDNA5 wave32, 16x16x32 bf16 -> f32) ----------
static __device__ __forceinline__ f32x8 wmma_bf16(bf16x16 a, bf16x16 b, f32x8 c) {
  return __builtin_amdgcn_wmma_f32_16x16x32_bf16(false, a, false, b, (short)0, c, false, false);
}
static __device__ __forceinline__ f32x8 zero8() {
  f32x8 c = {0.f,0.f,0.f,0.f,0.f,0.f,0.f,0.f}; return c;
}

// Fragment-order storage: one 16x32 (A) or 32x16 (B) tile = 32 lanes x 16 bf16,
// each lane's 16 values contiguous (32 B) -> single vector load per fragment.
static __device__ __forceinline__ bf16x16 ldFrag(const __bf16* base, int tileIdx, int lane) {
  return *(const bf16x16*)(base + ((size_t)tileIdx * 32 + lane) * 16);
}

// Swizzled offset of element (row, col) inside an A tile-row buffer
// (16 rows x K cols, K multiple of 32). Matches ISA 7.12.2 A layout.
static __device__ __forceinline__ int swzA(int row, int col) {
  int kt = col >> 5, kc = col & 31;
  int lane = (row & 15) | (((kc >> 3) & 1) << 4);
  int v = ((kc >> 4) << 2) | ((kc >> 1) & 3);
  int e = (v << 1) | (kc & 1);
  return (kt * 32 + lane) * 16 + e;
}

// ---------- weight swizzle: f32 row-major [K,N] -> bf16 fragment order ----------
__global__ void k_swzB(const float* __restrict__ W, __bf16* __restrict__ dst,
                       int K, int N, long srcStride, long dstStride) {
  int id = blockIdx.x * blockDim.x + threadIdx.x;
  int total = K * N;
  if (id >= total) return;
  int e = id & 15, lane = (id >> 4) & 31, tile = id >> 9;
  int ntiles = N >> 4;
  int nt = tile % ntiles, kt = tile / ntiles;
  int n = nt * 16 + (lane & 15);
  int half = lane >> 4, v = e >> 1, p = e & 1;
  int k = kt * 32 + ((e >> 3) << 4) + half * 8 + (e & 7);
  const float* src = W + (size_t)blockIdx.y * srcStride;
  dst[(size_t)blockIdx.y * dstStride + id] = f2bf(src[(size_t)k * N + n]);
}

// ---------- encoder: z0 = LN2(GELU(LN1(X0@ew1+eb1))@ew2+eb2) ----------
__global__ void __launch_bounds__(256) k_enc(
    const float* __restrict__ X,
    const float* __restrict__ eb1, const float* __restrict__ eg1, const float* __restrict__ ebt1,
    const float* __restrict__ eb2, const float* __restrict__ eg2, const float* __restrict__ ebt2,
    const __bf16* __restrict__ ew1s, const __bf16* __restrict__ ew2s,
    float* __restrict__ z, __bf16* __restrict__ zswz) {
  __shared__ __attribute__((aligned(32))) __bf16 sX[16 * 64];
  __shared__ float  sAct[16 * 128];
  __shared__ __attribute__((aligned(32))) __bf16 sActB[16 * 128];
  int t = threadIdx.x, lane = t & 31, wave = t >> 5;
  int b0 = blockIdx.x * 16;
  for (int i = t; i < 16 * 64; i += 256) {
    int r = i >> 6, cc = i & 63;
    sX[swzA(r, cc)] = f2bf(X[(size_t)(b0 + r) * (SEQ * DIN) + cc]);  // X[:,0,:]
  }
  __syncthreads();
  int n0 = wave * 16, half = lane >> 4, nn = lane & 15;
  f32x8 c0 = zero8(), c1 = zero8();
  c0 = wmma_bf16(ldFrag(sX, 0, lane), ldFrag(ew1s, 0 * 8 + wave, lane), c0);
  c1 = wmma_bf16(ldFrag(sX, 1, lane), ldFrag(ew1s, 1 * 8 + wave, lane), c1);
  f32x8 c = c0 + c1;
#pragma unroll
  for (int j = 0; j < 8; ++j) {
    int row = j + 8 * half, col = n0 + nn;
    sAct[row * 128 + col] = c[j] + eb1[col];
  }
  __syncthreads();
  { // LN1 + GELU (16 threads per row)
    int row = t >> 4, li = t & 15;
    float s = 0.f;
#pragma unroll
    for (int q = 0; q < 8; ++q) s += sAct[row * 128 + li + 16 * q];
    s = red16(s);
    float mean = s * (1.0f / 128.0f);
    float v = 0.f;
#pragma unroll
    for (int q = 0; q < 8; ++q) { float d = sAct[row * 128 + li + 16 * q] - mean; v += d * d; }
    v = red16(v);
    float rstd = rsqrtf(v * (1.0f / 128.0f) + 1e-5f);
#pragma unroll
    for (int q = 0; q < 8; ++q) {
      int col = li + 16 * q;
      float x = (sAct[row * 128 + col] - mean) * rstd * eg1[col] + ebt1[col];
      sActB[swzA(row, col)] = f2bf(gelu_erf(x));
    }
  }
  __syncthreads();
  c0 = zero8(); c1 = zero8();
#pragma unroll
  for (int kt = 0; kt < 4; kt += 2) {
    c0 = wmma_bf16(ldFrag(sActB, kt, lane), ldFrag(ew2s, kt * 8 + wave, lane), c0);
    c1 = wmma_bf16(ldFrag(sActB, kt + 1, lane), ldFrag(ew2s, (kt + 1) * 8 + wave, lane), c1);
  }
  c = c0 + c1;
  __syncthreads();
#pragma unroll
  for (int j = 0; j < 8; ++j) {
    int row = j + 8 * half, col = n0 + nn;
    sAct[row * 128 + col] = c[j] + eb2[col];
  }
  __syncthreads();
  { // LN2 -> z (fp32 state) + zswz (bf16 fragment order)
    int row = t >> 4, li = t & 15;
    float s = 0.f;
#pragma unroll
    for (int q = 0; q < 8; ++q) s += sAct[row * 128 + li + 16 * q];
    s = red16(s);
    float mean = s * (1.0f / 128.0f);
    float v = 0.f;
#pragma unroll
    for (int q = 0; q < 8; ++q) { float d = sAct[row * 128 + li + 16 * q] - mean; v += d * d; }
    v = red16(v);
    float rstd = rsqrtf(v * (1.0f / 128.0f) + 1e-5f);
#pragma unroll
    for (int q = 0; q < 8; ++q) {
      int col = li + 16 * q;
      float x = (sAct[row * 128 + col] - mean) * rstd * eg2[col] + ebt2[col];
      z[(size_t)(b0 + row) * DH + col] = x;
      zswz[(size_t)blockIdx.x * 2048 + swzA(row, col)] = f2bf(x);
    }
  }
}

// ---------- per-interval prep: dXdt and bias-contraction bterm ----------
__global__ void k_prep(const float* __restrict__ X, const float* __restrict__ ts,
                       const float* __restrict__ vb3, float* __restrict__ dX,
                       float* __restrict__ bterm, int s) {
  __shared__ float sdx[DIN];
  int b = blockIdx.x, t = threadIdx.x;
  float dtinv = 1.0f / (ts[s + 1] - ts[s]);
  if (t < DIN) {
    float d = (X[(size_t)b * (SEQ * DIN) + (s + 1) * DIN + t] -
               X[(size_t)b * (SEQ * DIN) + s * DIN + t]) * dtinv;
    sdx[t] = d;
    dX[(size_t)b * DIN + t] = d;
  }
  __syncthreads();
  float acc = 0.f;
#pragma unroll
  for (int i = 0; i < DIN; ++i) acc += vb3[t * DIN + i] * sdx[i];
  bterm[(size_t)b * DH + t] = acc;
}

// ---------- vector-field MLP ----------
__global__ void __launch_bounds__(512) k_vfmlp(
    const __bf16* __restrict__ zswz,
    const __bf16* __restrict__ vw1s, const float* __restrict__ vb1,
    const float* __restrict__ vg1, const float* __restrict__ vbt1,
    const __bf16* __restrict__ vw2s, const float* __restrict__ vb2,
    const float* __restrict__ vg2, const float* __restrict__ vbt2,
    __bf16* __restrict__ h2swz) {
  __shared__ __attribute__((aligned(32))) __bf16 sZ[16 * 128];
  __shared__ float  sA[16 * 256];
  __shared__ __attribute__((aligned(32))) __bf16 sAB[16 * 256];
  int t = threadIdx.x, lane = t & 31, wave = t >> 5;
  ((uint2*)sZ)[t] = ((const uint2*)(zswz + (size_t)blockIdx.x * 2048))[t];
  __syncthreads();
  int n0 = wave * 16, half = lane >> 4, nn = lane & 15;
  f32x8 c0 = zero8(), c1 = zero8();
#pragma unroll
  for (int kt = 0; kt < 4; kt += 2) {
    c0 = wmma_bf16(ldFrag(sZ, kt, lane), ldFrag(vw1s, kt * 16 + wave, lane), c0);
    c1 = wmma_bf16(ldFrag(sZ, kt + 1, lane), ldFrag(vw1s, (kt + 1) * 16 + wave, lane), c1);
  }
  f32x8 c = c0 + c1;
#pragma unroll
  for (int j = 0; j < 8; ++j) {
    int row = j + 8 * half, col = n0 + nn;
    sA[row * 256 + col] = c[j] + vb1[col];
  }
  __syncthreads();
  { // LN + GELU: wave per row
    int row = wave;
    float s = 0.f;
#pragma unroll
    for (int q = 0; q < 8; ++q) s += sA[row * 256 + lane + 32 * q];
    s = red32(s);
    float mean = s * (1.0f / 256.0f);
    float v = 0.f;
#pragma unroll
    for (int q = 0; q < 8; ++q) { float d = sA[row * 256 + lane + 32 * q] - mean; v += d * d; }
    v = red32(v);
    float rstd = rsqrtf(v * (1.0f / 256.0f) + 1e-5f);
#pragma unroll
    for (int q = 0; q < 8; ++q) {
      int col = lane + 32 * q;
      float x = (sA[row * 256 + col] - mean) * rstd * vg1[col] + vbt1[col];
      sAB[swzA(row, col)] = f2bf(gelu_erf(x));
    }
  }
  __syncthreads();
  c0 = zero8(); c1 = zero8();
#pragma unroll
  for (int kt = 0; kt < 8; kt += 2) {
    c0 = wmma_bf16(ldFrag(sAB, kt, lane), ldFrag(vw2s, kt * 16 + wave, lane), c0);
    c1 = wmma_bf16(ldFrag(sAB, kt + 1, lane), ldFrag(vw2s, (kt + 1) * 16 + wave, lane), c1);
  }
  c = c0 + c1;
  __syncthreads();
#pragma unroll
  for (int j = 0; j < 8; ++j) {
    int row = j + 8 * half, col = n0 + nn;
    sA[row * 256 + col] = c[j] + vb2[col];
  }
  __syncthreads();
  { // LN2 + GELU -> h2swz (contiguous 8 KB per 16-row tile)
    int row = wave;
    float s = 0.f;
#pragma unroll
    for (int q = 0; q < 8; ++q) s += sA[row * 256 + lane + 32 * q];
    s = red32(s);
    float mean = s * (1.0f / 256.0f);
    float v = 0.f;
#pragma unroll
    for (int q = 0; q < 8; ++q) { float d = sA[row * 256 + lane + 32 * q] - mean; v += d * d; }
    v = red32(v);
    float rstd = rsqrtf(v * (1.0f / 256.0f) + 1e-5f);
#pragma unroll
    for (int q = 0; q < 8; ++q) {
      int col = lane + 32 * q;
      float x = (sA[row * 256 + col] - mean) * rstd * vg2[col] + vbt2[col];
      h2swz[(size_t)blockIdx.x * 4096 + swzA(row, col)] = f2bf(gelu_erf(x));
    }
  }
}

// ---------- fused GEMM+contract ----------
__global__ void __launch_bounds__(128) k_contract(
    const __bf16* __restrict__ h2swz, const __bf16* __restrict__ vw3s,
    const float* __restrict__ dX, const float* __restrict__ bterm,
    float* __restrict__ dz) {
  __shared__ __attribute__((aligned(32))) __bf16 sH[16 * 256];  // 8 KB, fragment order
  __shared__ float  sPart[4][16];
  int t = threadIdx.x, lane = t & 31, wave = t >> 5;
  int b0 = blockIdx.x * 16, h = blockIdx.y;
  { // async copy: 8 KB contiguous tile (ASYNCcnt-tracked)
    const char* src = (const char*)(h2swz + (size_t)blockIdx.x * 4096);
    unsigned ldsbase = (unsigned)(uintptr_t)&sH[0];
#pragma unroll
    for (int r = 0; r < 4; ++r) {
      unsigned off = (unsigned)(r * 2048 + t * 16);
      unsigned ldsa = ldsbase + off;
      unsigned long long ga = (unsigned long long)(uintptr_t)(src + off);
      asm volatile("global_load_async_to_lds_b128 %0, %1, off"
                   :: "v"(ldsa), "v"(ga) : "memory");
    }
    asm volatile("s_wait_asynccnt 0x0" ::: "memory");
  }
  __syncthreads();
  int half = lane >> 4, nn = lane & 15;
  int ntB = h * 4 + wave;  // n-tile index into vw3s (ntiles = 512)
  f32x8 c0 = zero8(), c1 = zero8();
#pragma unroll
  for (int kt = 0; kt < 8; kt += 2) {
    if (kt + 2 < 8)
      __builtin_prefetch(vw3s + ((size_t)((kt + 2) * 512 + ntB) * 32) * 16, 0, 1);
    c0 = wmma_bf16(ldFrag(sH, kt, lane), ldFrag(vw3s, kt * 512 + ntB, lane), c0);
    c1 = wmma_bf16(ldFrag(sH, kt + 1, lane), ldFrag(vw3s, (kt + 1) * 512 + ntB, lane), c1);
  }
  f32x8 c = c0 + c1;
  int i_local = wave * 16 + nn;  // i within the 64-wide h-group
  float r[8];
#pragma unroll
  for (int j = 0; j < 8; ++j) {
    int b = b0 + j + 8 * half;
    float v = c[j] * dX[(size_t)b * DIN + i_local];
    r[j] = red16(v);
  }
  if (nn == 0) {
#pragma unroll
    for (int j = 0; j < 8; ++j) sPart[wave][j + 8 * half] = r[j];
  }
  __syncthreads();
  if (t < 16) {
    float s = sPart[0][t] + sPart[1][t] + sPart[2][t] + sPart[3][t] +
              bterm[(size_t)(b0 + t) * DH + h];
    dz[(size_t)(b0 + t) * DH + h] = s;
  }
}

// ---------- RK4 combine; maintains the swizzled bf16 vf-input copy ----------
__global__ void k_rk(float* __restrict__ z, const float* __restrict__ dz,
                     float* __restrict__ zacc, __bf16* __restrict__ zswz,
                     const float* __restrict__ ts, int s, int stage) {
  int i = blockIdx.x * blockDim.x + threadIdx.x;
  if (i >= BATCH * DH) return;
  float dt = (ts[s + 1] - ts[s]) * (1.0f / (float)NSUB);
  float d = dz[i];
  float w;
  if (stage == 0)      { zacc[i] = d;          w = z[i] + 0.5f * dt * d; }
  else if (stage == 1) { zacc[i] += 2.0f * d;  w = z[i] + 0.5f * dt * d; }
  else if (stage == 2) { zacc[i] += 2.0f * d;  w = z[i] + dt * d; }
  else { w = z[i] + (dt * (1.0f / 6.0f)) * (zacc[i] + d); z[i] = w; }
  int b = i >> 7, col = i & 127;
  zswz[(size_t)(b >> 4) * 2048 + swzA(b & 15, col)] = f2bf(w);
}

// ---------- attention-equivalent ----------
__global__ void __launch_bounds__(256) k_attn(
    const __bf16* __restrict__ zswz, const __bf16* __restrict__ wvs, const float* __restrict__ bv,
    const __bf16* __restrict__ wos, const float* __restrict__ bo,
    __bf16* __restrict__ attswz) {
  __shared__ __attribute__((aligned(32))) __bf16 sZ[16 * 128];
  __shared__ float  sT[16 * 128];
  __shared__ __attribute__((aligned(32))) __bf16 sTB[16 * 128];
  int t = threadIdx.x, lane = t & 31, wave = t >> 5;
  ((uint4*)sZ)[t] = ((const uint4*)(zswz + (size_t)blockIdx.x * 2048))[t];
  __syncthreads();
  int n0 = wave * 16, half = lane >> 4, nn = lane & 15;
  f32x8 c0 = zero8(), c1 = zero8();
#pragma unroll
  for (int kt = 0; kt < 4; kt += 2) {
    c0 = wmma_bf16(ldFrag(sZ, kt, lane), ldFrag(wvs, kt * 8 + wave, lane), c0);
    c1 = wmma_bf16(ldFrag(sZ, kt + 1, lane), ldFrag(wvs, (kt + 1) * 8 + wave, lane), c1);
  }
  f32x8 c = c0 + c1;
#pragma unroll
  for (int j = 0; j < 8; ++j) {
    int row = j + 8 * half, col = n0 + nn;
    sT[row * 128 + col] = c[j] + bv[col];
  }
  __syncthreads();
  for (int i = t; i < 16 * 128; i += 256) {
    int row = i >> 7, col = i & 127;
    sTB[swzA(row, col)] = f2bf(sT[i]);
  }
  __syncthreads();
  c0 = zero8(); c1 = zero8();
#pragma unroll
  for (int kt = 0; kt < 4; kt += 2) {
    c0 = wmma_bf16(ldFrag(sTB, kt, lane), ldFrag(wos, kt * 8 + wave, lane), c0);
    c1 = wmma_bf16(ldFrag(sTB, kt + 1, lane), ldFrag(wos, (kt + 1) * 8 + wave, lane), c1);
  }
  c = c0 + c1;
#pragma unroll
  for (int j = 0; j < 8; ++j) {
    int row = j + 8 * half, col = n0 + nn;
    attswz[(size_t)blockIdx.x * 2048 + swzA(row, col)] = f2bf(c[j] + bo[col]);
  }
}

// ---------- trait decoders ----------
__global__ void __launch_bounds__(128) k_dec1(
    const __bf16* __restrict__ attswz, const __bf16* __restrict__ dw1s,
    const float* __restrict__ db1, __bf16* __restrict__ d1swz) {
  int t = threadIdx.x, lane = t & 31, wave = t >> 5;
  int k = blockIdx.y;
  const __bf16* attA = attswz + (size_t)blockIdx.x * 2048;
  const __bf16* W = dw1s + (size_t)k * (DH * 64);
  int half = lane >> 4, nn = lane & 15, n0 = wave * 16;
  f32x8 c0 = zero8(), c1 = zero8();
#pragma unroll
  for (int kt = 0; kt < 4; kt += 2) {
    c0 = wmma_bf16(ldFrag(attA, kt, lane), ldFrag(W, kt * 4 + wave, lane), c0);
    c1 = wmma_bf16(ldFrag(attA, kt + 1, lane), ldFrag(W, (kt + 1) * 4 + wave, lane), c1);
  }
  f32x8 c = c0 + c1;
#pragma unroll
  for (int j = 0; j < 8; ++j) {
    int row = j + 8 * half, col = n0 + nn;
    float x = c[j] + db1[k * 64 + col];
    d1swz[(size_t)k * (BATCH * 64) + (size_t)blockIdx.x * 1024 + swzA(row, col)] =
        f2bf(gelu_erf(x));
  }
}
__global__ void __launch_bounds__(64) k_dec2(
    const __bf16* __restrict__ d1swz, const __bf16* __restrict__ dw2s,
    const float* __restrict__ db2, __bf16* __restrict__ d2b) {
  int t = threadIdx.x, lane = t & 31, wave = t >> 5;
  int k = blockIdx.y;
  const __bf16* A = d1swz + (size_t)k * (BATCH * 64) + (size_t)blockIdx.x * 1024;
  const __bf16* W = dw2s + (size_t)k * (64 * 32);
  int half = lane >> 4, nn = lane & 15, n0 = wave * 16;
  f32x8 c0 = zero8(), c1 = zero8();
  c0 = wmma_bf16(ldFrag(A, 0, lane), ldFrag(W, 0 * 2 + wave, lane), c0);
  c1 = wmma_bf16(ldFrag(A, 1, lane), ldFrag(W, 1 * 2 + wave, lane), c1);
  f32x8 c = c0 + c1;
#pragma unroll
  for (int j = 0; j < 8; ++j) {
    int row = j + 8 * half, col = n0 + nn;
    float x = c[j] + db2[k * 32 + col];
    d2b[(size_t)k * (BATCH * 32) + (size_t)(blockIdx.x * 16 + row) * 32 + col] =
        f2bf(gelu_erf(x));
  }
}
__global__ void k_dec3(const __bf16* __restrict__ d2b, const float* __restrict__ dw3,
                       const float* __restrict__ db3, float* __restrict__ out) {
  int id = blockIdx.x * blockDim.x + threadIdx.x;
  if (id >= BATCH * NTRAITS) return;
  int b = id / NTRAITS, k = id % NTRAITS;
  float s = db3[k];
#pragma unroll
  for (int e = 0; e < 32; ++e)
    s += bf2f(d2b[(size_t)k * (BATCH * 32) + (size_t)b * 32 + e]) * dw3[k * 32 + e];
  out[(size_t)b * NTRAITS + k] = 1.0f / (1.0f + expf(-s));
}

// ---------- host orchestration ----------
extern "C" void kernel_launch(void* const* d_in, const int* in_sizes, int n_in,
                              void* d_out, int out_size, void* d_ws, size_t ws_size,
                              hipStream_t stream) {
  (void)in_sizes; (void)n_in; (void)out_size; (void)ws_size;
  const float* X    = (const float*)d_in[0];
  const float* ts   = (const float*)d_in[1];
  const float* ew1  = (const float*)d_in[2];
  const float* eb1  = (const float*)d_in[3];
  const float* eg1  = (const float*)d_in[4];
  const float* ebt1 = (const float*)d_in[5];
  const float* ew2  = (const float*)d_in[6];
  const float* eb2  = (const float*)d_in[7];
  const float* eg2  = (const float*)d_in[8];
  const float* ebt2 = (const float*)d_in[9];
  const float* vw1  = (const float*)d_in[10];
  const float* vb1  = (const float*)d_in[11];
  const float* vg1  = (const float*)d_in[12];
  const float* vbt1 = (const float*)d_in[13];
  const float* vw2  = (const float*)d_in[14];
  const float* vb2  = (const float*)d_in[15];
  const float* vg2  = (const float*)d_in[16];
  const float* vbt2 = (const float*)d_in[17];
  const float* vw3  = (const float*)d_in[18];
  const float* vb3  = (const float*)d_in[19];
  const float* wv   = (const float*)d_in[20];
  const float* bv   = (const float*)d_in[21];
  const float* wo   = (const float*)d_in[22];
  const float* bo   = (const float*)d_in[23];
  const float* dw1  = (const float*)d_in[24];
  const float* db1  = (const float*)d_in[25];
  const float* dw2  = (const float*)d_in[26];
  const float* db2  = (const float*)d_in[27];
  const float* dw3  = (const float*)d_in[28];
  const float* db3  = (const float*)d_in[29];

  char* base = (char*)d_ws; size_t off = 0;
  auto alloc = [&](size_t bytes) -> void* {
    void* p = base + off;
    off = (off + bytes + 255) & ~(size_t)255;
    return p;
  };
  __bf16* vw1s = (__bf16*)alloc((size_t)DH  * DH2 * 2);
  __bf16* vw2s = (__bf16*)alloc((size_t)DH2 * DH2 * 2);
  __bf16* vw3s = (__bf16*)alloc((size_t)DH2 * DHI * 2);
  __bf16* ew1s = (__bf16*)alloc((size_t)DIN * DH  * 2);
  __bf16* ew2s = (__bf16*)alloc((size_t)DH  * DH  * 2);
  __bf16* wvs  = (__bf16*)alloc((size_t)DH  * DH  * 2);
  __bf16* wos  = (__bf16*)alloc((size_t)DH  * DH  * 2);
  __bf16* dw1s = (__bf16*)alloc((size_t)NTRAITS * DH * 64 * 2);
  __bf16* dw2s = (__bf16*)alloc((size_t)NTRAITS * 64 * 32 * 2);
  float*  z    = (float*) alloc((size_t)BATCH * DH  * 4);
  float*  zacc = (float*) alloc((size_t)BATCH * DH  * 4);
  float*  dz   = (float*) alloc((size_t)BATCH * DH  * 4);
  __bf16* zswz = (__bf16*)alloc((size_t)BATCH * DH  * 2);
  __bf16* h2swz= (__bf16*)alloc((size_t)BATCH * DH2 * 2);
  float*  dXb  = (float*) alloc((size_t)BATCH * DIN * 4);
  float*  bterm= (float*) alloc((size_t)BATCH * DH  * 4);
  __bf16* attswz=(__bf16*)alloc((size_t)BATCH * DH  * 2);
  __bf16* d1swz= (__bf16*)alloc((size_t)NTRAITS * BATCH * 64 * 2);
  __bf16* d2b  = (__bf16*)alloc((size_t)NTRAITS * BATCH * 32 * 2);

  auto swz = [&](const float* W, __bf16* dst, int K, int N, int batch,
                 long sstr, long dstr) {
    dim3 g((K * N + 255) / 256, batch);
    k_swzB<<<g, 256, 0, stream>>>(W, dst, K, N, sstr, dstr);
  };
  swz(vw1, vw1s, DH,  DH2, 1, 0, 0);
  swz(vw2, vw2s, DH2, DH2, 1, 0, 0);
  swz(vw3, vw3s, DH2, DHI, 1, 0, 0);
  swz(ew1, ew1s, DIN, DH,  1, 0, 0);
  swz(ew2, ew2s, DH,  DH,  1, 0, 0);
  swz(wv,  wvs,  DH,  DH,  1, 0, 0);
  swz(wo,  wos,  DH,  DH,  1, 0, 0);
  swz(dw1, dw1s, DH, 64, NTRAITS, (long)DH * 64, (long)DH * 64);
  swz(dw2, dw2s, 64, 32, NTRAITS, (long)64 * 32, (long)64 * 32);

  k_enc<<<BATCH / 16, 256, 0, stream>>>(X, eb1, eg1, ebt1, eb2, eg2, ebt2,
                                        ew1s, ew2s, z, zswz);

  for (int s = 0; s < SEQ - 1; ++s) {
    k_prep<<<BATCH, 128, 0, stream>>>(X, ts, vb3, dXb, bterm, s);
    for (int sub = 0; sub < NSUB; ++sub) {
      for (int st = 0; st < 4; ++st) {
        k_vfmlp<<<BATCH / 16, 512, 0, stream>>>(zswz, vw1s, vb1, vg1, vbt1,
                                                vw2s, vb2, vg2, vbt2, h2swz);
        k_contract<<<dim3(BATCH / 16, DH), 128, 0, stream>>>(h2swz, vw3s, dXb, bterm, dz);
        k_rk<<<(BATCH * DH + 255) / 256, 256, 0, stream>>>(z, dz, zacc, zswz, ts, s, st);
      }
    }
  }

  k_attn<<<BATCH / 16, 256, 0, stream>>>(zswz, wvs, bv, wos, bo, attswz);
  k_dec1<<<dim3(BATCH / 16, NTRAITS), 128, 0, stream>>>(attswz, dw1s, db1, d1swz);
  k_dec2<<<dim3(BATCH / 16, NTRAITS), 64, 0, stream>>>(d1swz, dw2s, db2, d2b);
  k_dec3<<<(BATCH * NTRAITS + 255) / 256, 256, 0, stream>>>(d2b, dw3, db3, (float*)d_out);
}
